// WindowAttention_36790689857810
// MI455X (gfx1250) — compile-verified, matrix operand order checked by reading, not run on hardware
//
#include <hip/hip_runtime.h>

typedef __attribute__((ext_vector_type(16))) __bf16 v16bf;
typedef __attribute__((ext_vector_type(8)))  __bf16 bf16x8;
typedef __attribute__((ext_vector_type(8)))  float  v8f;
typedef __attribute__((ext_vector_type(4)))  float  f32x4;

#define NTOK 49
#define NP   64
#define CDIM 128
#define HEADS 4
#define HD   32

// LDS strides (elements): multiples of 8 so every operand fetch is a 16B ds_load_b128.
// Byte strides 272/80/144 -> dword steps 68/20/36: distinct 4-dword chunks over 16 lanes.
#define XS   136          // x rows (K=128)
#define QKS  40           // q,k rows (K=32)
#define QKH  (64*QKS)     // per-head q/k block (2560)
#define VTS  72           // v^T rows (64 m-values)
#define VTH  (32*VTS)     // 2304
#define PBS  72           // softmax probs rows
#define PBH  (64*PBS)     // 4608
#define OBS  72           // per-pair attention output (64 cols used)

__device__ __forceinline__ v8f wmma_bf16(v16bf a, v16bf b, v8f c) {
  return __builtin_amdgcn_wmma_f32_16x16x32_bf16(false, a, false, b,
                                                 (short)0, c, false, false);
}

// 16x32 bf16 operand tile in WMMA A layout (lane&15 = row; per lane the K-run is
// 8 contiguous bf16 per half-tile -> two ds_load_b128). Also used for B operands
// whose source is stored transposed (16-dim as rows).
__device__ __forceinline__ v16bf lds_tileA(const __bf16* base, int stride,
                                           int row0, int col0, int lane) {
  const int half = lane >> 4;
  const __bf16* rp = base + (row0 + (lane & 15)) * stride + col0 + (half << 3);
  bf16x8 lo = *(const bf16x8*)(rp);        // k = kb+0..7   -> a[0..7]
  bf16x8 hi = *(const bf16x8*)(rp + 16);   // k = kb+16..23 -> a[8..15]
  return __builtin_shufflevector(lo, hi, 0, 1, 2, 3, 4, 5, 6, 7,
                                 8, 9, 10, 11, 12, 13, 14, 15);
}

// B tile (32 x 16) from a row-major f32 weight matrix W[n][k]
// (X @ W^T -> B(k,n) = W[n][k]); four global_load_b128 + cvt to bf16.
__device__ __forceinline__ v16bf glb_tileB(const float* __restrict__ W, int ld,
                                           int n0, int k0, int lane) {
  const int half = lane >> 4;
  const float* rp = W + (n0 + (lane & 15)) * ld + k0 + (half << 3);
  f32x4 p0 = *(const f32x4*)(rp);
  f32x4 p1 = *(const f32x4*)(rp + 4);
  f32x4 p2 = *(const f32x4*)(rp + 16);
  f32x4 p3 = *(const f32x4*)(rp + 20);
  v16bf b;
  b[0]  = (__bf16)p0.x; b[1]  = (__bf16)p0.y; b[2]  = (__bf16)p0.z; b[3]  = (__bf16)p0.w;
  b[4]  = (__bf16)p1.x; b[5]  = (__bf16)p1.y; b[6]  = (__bf16)p1.z; b[7]  = (__bf16)p1.w;
  b[8]  = (__bf16)p2.x; b[9]  = (__bf16)p2.y; b[10] = (__bf16)p2.z; b[11] = (__bf16)p2.w;
  b[12] = (__bf16)p3.x; b[13] = (__bf16)p3.y; b[14] = (__bf16)p3.z; b[15] = (__bf16)p3.w;
  return b;
}

__global__ __launch_bounds__(256)
void win_attn_fused(const float* __restrict__ x,
                    const float* __restrict__ maskp,
                    const float* __restrict__ qkv_w,
                    const float* __restrict__ qkv_b,
                    const float* __restrict__ proj_w,
                    const float* __restrict__ proj_b,
                    const float* __restrict__ bias_table,
                    float* __restrict__ outp) {
  __shared__ __align__(16) __bf16 xb[NP * XS];        // window input, bf16
  __shared__ __align__(16) __bf16 u_qkpb[10240];      // {q|k} pair UNION probs pair
  __shared__ __align__(16) __bf16 vT2[2 * VTH];       // v^T per head (pair)
  __shared__ __align__(16) __bf16 obp[NP * OBS];      // attn output (pair)
  __bf16* qs2 = u_qkpb;            // [2][64][QKS]
  __bf16* ks2 = u_qkpb + 2 * QKH;  // [2][64][QKS]
  __bf16* pb2 = u_qkpb;            // [2][64][PBS]  (q/k dead when probs written)

  const int tid  = threadIdx.x;
  const int wv   = tid >> 5;
  const int lane = tid & 31;
  const int ln   = lane & 15;
  const int half = lane >> 4;
  const int bwin = blockIdx.x;
  const int wmsk = bwin & 63;      // window index into mask (B_ = groups*64)

  // ---- stage 0: x window -> LDS bf16 (pad rows 49..63 with zeros) ----
  for (int i = tid; i < NP * (CDIM / 8); i += 256) {
    int m  = i >> 4;
    int c8 = (i & 15) << 3;
    f32x4 t0 = {0.f, 0.f, 0.f, 0.f}, t1 = {0.f, 0.f, 0.f, 0.f};
    if (m < NTOK) {
      const float* xp = x + ((size_t)bwin * NTOK + m) * CDIM + c8;
      t0 = *(const f32x4*)(xp);
      t1 = *(const f32x4*)(xp + 4);
    }
    bf16x8 h8;
    h8[0] = (__bf16)t0.x; h8[1] = (__bf16)t0.y; h8[2] = (__bf16)t0.z; h8[3] = (__bf16)t0.w;
    h8[4] = (__bf16)t1.x; h8[5] = (__bf16)t1.y; h8[6] = (__bf16)t1.z; h8[7] = (__bf16)t1.w;
    *(bf16x8*)(xb + m * XS + c8) = h8;
  }

  const v8f zf = {};
  v8f pacc[4];                     // projection accumulators, live across pairs
#pragma unroll
  for (int i = 0; i < 4; ++i) pacc[i] = zf;

  const float SCALE = 0.17677669529663687f;   // 32^-0.5 folded into q

#pragma unroll 1
  for (int p = 0; p < 2; ++p) {    // head pairs {0,1}, {2,3}
    __syncthreads();               // protects u_qkpb/obp reuse (+stage0 on p==0)

    // ---- stage 1: qkv GEMM for this pair's 192 columns (q|k|v of 2 heads) ----
    for (int t = wv; t < 48; t += 8) {
      int m0  = (t & 3) << 4;
      int lc0 = (t >> 2) << 4;
      int sub = lc0 >> 6;                       // 0=q 1=k 2=v
      int hh  = (lc0 & 63) >> 5;                // head within pair
      int d0  = lc0 & 31;                       // 0 or 16
      int gn0 = sub * CDIM + ((p << 1) + hh) * HD + d0;
      v8f acc = zf;
#pragma unroll
      for (int kt = 0; kt < 4; ++kt) {
        v16bf a  = lds_tileA(xb, XS, m0, kt * 32, lane);
        v16bf bm = glb_tileB(qkv_w, CDIM, gn0, kt * 32, lane);
        acc = wmma_bf16(a, bm, acc);
      }
      float bia = qkv_b[gn0 + ln];
      int d = d0 + ln;
#pragma unroll
      for (int r = 0; r < 8; ++r) {             // D: VGPR r -> M = r + 8*half
        int m = m0 + r + (half << 3);
        float val = acc[r] + bia;
        if (sub == 0)      qs2[hh * QKH + m * QKS + d] = (__bf16)(val * SCALE);
        else if (sub == 1) ks2[hh * QKH + m * QKS + d] = (__bf16)val;
        else               vT2[hh * VTH + d * VTS + m] = (__bf16)val;
      }
    }
    __syncthreads();

    // ---- stage 2: scores + in-register softmax (wave owns 16 rows of 1 head) ----
    const int hh = wv >> 2;
    const int h  = (p << 1) + hh;
    const int m0 = (wv & 3) << 4;
    v16bf aq = lds_tileA(qs2 + hh * QKH, QKS, m0, 0, lane);
    v8f sacc[4];
#pragma unroll
    for (int nt = 0; nt < 4; ++nt) {
      v16bf bk = lds_tileA(ks2 + hh * QKH, QKS, nt << 4, 0, lane);
      sacc[nt] = wmma_bf16(aq, bk, zf);
    }
    float pr[4][8];
#pragma unroll
    for (int r = 0; r < 8; ++r) {
      int m = m0 + r + (half << 3);
      int mh = m / 7, mw = m % 7;
      float sv[4];
#pragma unroll
      for (int nt = 0; nt < 4; ++nt) {
        int n = (nt << 4) + ln;
        float s = -1e30f;
        if (m < NTOK && n < NTOK) {
          int nh = n / 7, nw = n % 7;
          int ridx = (mh - nh + 6) * 13 + (mw - nw + 6);
          s = sacc[nt][r] + bias_table[ridx * HEADS + h]
              + maskp[(wmsk * NTOK + m) * NTOK + n];
        }
        sv[nt] = s;
      }
      float mx = fmaxf(fmaxf(sv[0], sv[1]), fmaxf(sv[2], sv[3]));
#pragma unroll
      for (int off = 1; off < 16; off <<= 1)
        mx = fmaxf(mx, __shfl_xor(mx, off, 32));
      float sum = 0.f;
#pragma unroll
      for (int nt = 0; nt < 4; ++nt) { sv[nt] = __expf(sv[nt] - mx); sum += sv[nt]; }
#pragma unroll
      for (int off = 1; off < 16; off <<= 1)
        sum += __shfl_xor(sum, off, 32);
      float inv = __builtin_amdgcn_rcpf(sum);
#pragma unroll
      for (int nt = 0; nt < 4; ++nt) pr[nt][r] = sv[nt] * inv;
    }
    __syncthreads();               // all waves finished reading q/k (aliased w/ pb)
#pragma unroll
    for (int r = 0; r < 8; ++r) {
      int m = m0 + r + (half << 3);
#pragma unroll
      for (int nt = 0; nt < 4; ++nt)
        pb2[hh * PBH + m * PBS + (nt << 4) + ln] = (__bf16)pr[nt][r];
    }

    // ---- attn @ v: same wave consumes exactly the pb rows it wrote ----
#pragma unroll
    for (int nt2 = 0; nt2 < 2; ++nt2) {
      v8f acc = zf;
#pragma unroll
      for (int kt = 0; kt < 2; ++kt) {
        v16bf a  = lds_tileA(pb2 + hh * PBH, PBS, m0,       kt * 32, lane);
        v16bf bv = lds_tileA(vT2 + hh * VTH, VTS, nt2 << 4, kt * 32, lane);
        acc = wmma_bf16(a, bv, acc);
      }
#pragma unroll
      for (int r = 0; r < 8; ++r) {
        int m = m0 + r + (half << 3);
        obp[m * OBS + hh * HD + (nt2 << 4) + ln] = (__bf16)acc[r];
      }
    }
    __syncthreads();

    // ---- stage 3: projection partial (K = this pair's 64 channels) ----
#pragma unroll
    for (int i = 0; i < 4; ++i) {
      int t   = wv + (i << 3);
      int mt0 = (t & 3) << 4;
      int nt0 = (t >> 2) << 4;
#pragma unroll
      for (int kt = 0; kt < 2; ++kt) {
        v16bf a  = lds_tileA(obp, OBS, mt0, kt * 32, lane);
        v16bf bw = glb_tileB(proj_w, CDIM, nt0, (p << 6) + kt * 32, lane);
        pacc[i] = wmma_bf16(a, bw, pacc[i]);
      }
    }
  }

  // ---- final writeback: out = proj + proj_b (valid rows only) ----
#pragma unroll
  for (int i = 0; i < 4; ++i) {
    int t   = wv + (i << 3);
    int mt0 = (t & 3) << 4;
    int n   = ((t >> 2) << 4) + ln;
    float pbv = proj_b[n];
#pragma unroll
    for (int r = 0; r < 8; ++r) {
      int m = mt0 + r + (half << 3);
      if (m < NTOK)
        outp[((size_t)bwin * NTOK + m) * CDIM + n] = pacc[i][r] + pbv;
    }
  }
}

extern "C" void kernel_launch(void* const* d_in, const int* in_sizes, int n_in,
                              void* d_out, int out_size, void* d_ws, size_t ws_size,
                              hipStream_t stream) {
  const float* x          = (const float*)d_in[0];
  const float* mask       = (const float*)d_in[1];
  const float* qkv_w      = (const float*)d_in[2];
  const float* qkv_b      = (const float*)d_in[3];
  const float* proj_w     = (const float*)d_in[4];
  const float* proj_b     = (const float*)d_in[5];
  const float* bias_table = (const float*)d_in[6];
  float* out = (float*)d_out;

  const int B_ = in_sizes[0] / (NTOK * CDIM);   // 8192 windows
  win_attn_fused<<<dim3(B_), dim3(256), 0, stream>>>(
      x, mask, qkv_w, qkv_b, proj_w, proj_b, bias_table, out);
}
